// MultiHeadAttention_86732569575433
// MI455X (gfx1250) — matrix operand order repaired and checked by reading, not compiled, not run on hardware
//
#include <hip/hip_runtime.h>
#include <hip/hip_bf16.h>
#include <stdint.h>

// ---------------------------------------------------------------------------
// MI455X (gfx1250) multi-head attention forward.
// bf16 WMMA (v_wmma_f32_16x16x32_bf16) + fp32 accumulate everywhere;
// TDM (tensor_load_to_lds) stages K tiles into LDS with row padding chosen
// so B-fragment ds_load_b128 reads are bank-conflict-free.
// ---------------------------------------------------------------------------

typedef __attribute__((ext_vector_type(16))) __bf16 v16bf;
typedef __attribute__((ext_vector_type(8)))  float  v8f;
typedef __attribute__((ext_vector_type(4)))  unsigned int u32x4;
typedef __attribute__((ext_vector_type(4)))  int i32x4;
typedef __attribute__((ext_vector_type(8)))  int i32x8;

union Frag16 {            // one WMMA 16-bit A or B fragment: 16 bf16 = 8 VGPRs
    v16bf v;
    uint4 q[2];
};

#define WMMA_BF16(A, B, C) \
    __builtin_amdgcn_wmma_f32_16x16x32_bf16(false, (A), false, (B), (short)0, (C), false, false)

#if defined(__has_builtin)
#if __has_builtin(__builtin_amdgcn_tensor_load_to_lds)
#define ATHENA_HAVE_TDM 1
#endif
#endif
#ifndef ATHENA_HAVE_TDM
#define ATHENA_HAVE_TDM 0
#endif

__device__ __forceinline__ unsigned short f32_to_bf16(float f) {
    unsigned int u = __float_as_uint(f);
    unsigned int r = (u + 0x7FFFu + ((u >> 16) & 1u)) >> 16;   // RNE
    return (unsigned short)r;
}

__device__ __forceinline__ float half16_max(float v) {
    v = fmaxf(v, __shfl_xor(v, 1, 32));
    v = fmaxf(v, __shfl_xor(v, 2, 32));
    v = fmaxf(v, __shfl_xor(v, 4, 32));
    v = fmaxf(v, __shfl_xor(v, 8, 32));
    return v;
}
__device__ __forceinline__ float half16_sum(float v) {
    v += __shfl_xor(v, 1, 32);
    v += __shfl_xor(v, 2, 32);
    v += __shfl_xor(v, 4, 32);
    v += __shfl_xor(v, 8, 32);
    return v;
}

// ---------------------------------------------------------------------------
// TDM: issue a 2D tile load (16 rows x 64 bf16) global -> LDS.
// Pad config: 4 dwords after every 32-dword (128B) row -> LDS row stride
// 144B = 36 dwords; 36*lr mod 64 covers all 16 disjoint 4-bank groups, so the
// per-lane ds_load_b128 B-fragment reads are conflict-free.
// ---------------------------------------------------------------------------
#define TDM_PAD_CFG ((1u << 20) | (4u << 22) | (3u << 25)) // pad_en, int=32dw, amt=4dw
#define KROW 72   // padded LDS row stride in bf16 elements (144 bytes)

#if ATHENA_HAVE_TDM
__device__ __forceinline__ void tdm_issue_k_tile(const unsigned short* gsrc,
                                                 unsigned lds_byte_off) {
    unsigned long long ga = (unsigned long long)(uintptr_t)gsrc;
    u32x4 g0;
    g0[0] = 1u;                                         // count=1 (valid user D#)
    g0[1] = lds_byte_off;                               // lds_addr
    g0[2] = (unsigned)(ga & 0xFFFFFFFFu);               // global_addr[31:0]
    g0[3] = (unsigned)((ga >> 32) & 0x01FFFFFFu) | (2u << 30);  // addr[56:32]|type=2
    i32x8 g1;
    g1[0] = (int)((1u << 16) | TDM_PAD_CFG);            // data_size=2B + pad cfg
    g1[1] = (int)(64u << 16);                           // tensor_dim0 = 64
    g1[2] = (int)(16u << 16);                           // tensor_dim1 = 16
    g1[3] = (int)(64u << 16);                           // tile_dim0 = 64
    g1[4] = (int)16;                                    // tile_dim1 = 16
    g1[5] = (int)1024;                                  // tensor_dim0_stride = 1024
    g1[6] = 0;
    g1[7] = 0;
    i32x4 z4; z4[0] = z4[1] = z4[2] = z4[3] = 0;
#if defined(__clang_major__) && (__clang_major__ >= 23)
    i32x8 z8;
    z8[0]=z8[1]=z8[2]=z8[3]=z8[4]=z8[5]=z8[6]=z8[7]=0;
    __builtin_amdgcn_tensor_load_to_lds(g0, g1, z4, z4, z8, 0);
#else
    __builtin_amdgcn_tensor_load_to_lds(g0, g1, z4, z4, 0);
#endif
}
#endif

template <int N>
__device__ __forceinline__ void tdm_wait() {
#if ATHENA_HAVE_TDM
    __builtin_amdgcn_s_wait_tensorcnt((short)N);
#endif
}

// Fallback staging (same padded layout) if the TDM builtin is unavailable.
__device__ __forceinline__ void stage_k_tile_coop(unsigned short* dst,
                                                  const unsigned short* src,
                                                  int tid) {
    for (int idx = tid; idx < 16 * 64; idx += 128) {
        const int r = idx >> 6, c = idx & 63;
        dst[r * KROW + c] = src[(size_t)r * 1024 + c];
    }
}

// ---------------------------------------------------------------------------
// Kernel 0: fp32 -> bf16 convert
// ---------------------------------------------------------------------------
__global__ void cvt_bf16_kernel(const float* __restrict__ src,
                                unsigned short* __restrict__ dst, int n) {
    int i = blockIdx.x * blockDim.x + threadIdx.x;
    if (i < n) dst[i] = f32_to_bf16(src[i]);
}

// ---------------------------------------------------------------------------
// Kernel 1: projection GEMM  Y[4096,1024] = X[4096,1024] @ W^T + bias.
// Wave computes a 16(M) x 64(N) strip.  A/B fragments are register
// double-buffered: next K-step loads issue while current WMMAs run, so the
// compiler doesn't have to s_wait_loadcnt 0 before every WMMA.
// vt_mode==1 stores V transposed as Vt[b*16+h][d][key] for the P@V GEMM.
// ---------------------------------------------------------------------------
__global__ void __launch_bounds__(128)
proj_gemm_kernel(const unsigned short* __restrict__ X,
                 const unsigned short* __restrict__ W,
                 const float* __restrict__ bias,
                 unsigned short* __restrict__ Y,
                 int vt_mode)
{
    const int lane = threadIdx.x & 31;
    const int widx = threadIdx.x >> 5;
    const int lr   = lane & 15;
    const int hi   = lane >> 4;
    const int m0   = (blockIdx.x >> 2) << 4;                    // 256 row tiles
    const int n0   = (((blockIdx.x & 3) << 2) + widx) << 6;     // 16 strips of 64

    v8f acc[4];
    #pragma unroll
    for (int j = 0; j < 4; ++j)
        #pragma unroll
        for (int e = 0; e < 8; ++e) acc[j][e] = 0.0f;

    const unsigned short* arow = X + (size_t)(m0 + lr) * 1024;

    Frag16 a[2], b[2][4];
    auto loadA = [&](int buf, int kk) {
        a[buf].q[0] = *(const uint4*)(arow + kk + hi * 16);
        a[buf].q[1] = *(const uint4*)(arow + kk + hi * 16 + 8);
    };
    auto loadB = [&](int buf, int kk) {
        #pragma unroll
        for (int j = 0; j < 4; ++j) {
            const unsigned short* brow =
                W + (size_t)(n0 + j * 16 + lr) * 1024 + kk + hi * 16;
            b[buf][j].q[0] = *(const uint4*)(brow);
            b[buf][j].q[1] = *(const uint4*)(brow + 8);
        }
    };

    loadA(0, 0);
    loadB(0, 0);
    #pragma unroll 2
    for (int kk = 0; kk < 1024; kk += 32) {
        const int cur = (kk >> 5) & 1;
        if (kk + 32 < 1024) {          // software pipeline: prefetch next stage
            loadA(cur ^ 1, kk + 32);
            loadB(cur ^ 1, kk + 32);
        }
        #pragma unroll
        for (int j = 0; j < 4; ++j)
            acc[j] = WMMA_BF16(a[cur].v, b[cur][j].v, acc[j]);
    }

    #pragma unroll
    for (int j = 0; j < 4; ++j) {
        const int col = n0 + j * 16 + lr;
        const float bv = bias[col];
        #pragma unroll
        for (int i = 0; i < 8; ++i) {
            const int row = m0 + i + hi * 8;    // C layout: VGPR i -> M=i(+8)
            const unsigned short hval = f32_to_bf16(acc[j][i] + bv);
            if (vt_mode == 0) {
                Y[(size_t)row * 1024 + col] = hval;
            } else {                            // V transposed: [b*16+h][d][key]
                const int bb = row >> 11, key = row & 2047;
                const int head = col >> 6, d = col & 63;
                Y[((size_t)((bb * 16 + head) * 64 + d)) * 2048 + key] = hval;
            }
        }
    }
}

// ---------------------------------------------------------------------------
// Kernel 2: attention.  One block = (b, h, 64 q-rows); 4 waves x 16 rows.
// K tiles (16 keys x 64 d) are staged global->LDS by the Tensor Data Mover,
// double-buffered on TENSORcnt; all 4 waves consume them from LDS.
// Pass 1: online row max/sum.  Pass 2: fp32 probs to d_out + P@V via LDS
// restage of bf16 probs as A fragments.
// ---------------------------------------------------------------------------
__global__ void __launch_bounds__(128)
attn_kernel(const unsigned short* __restrict__ Qh,   // [4096][1024]
            const unsigned short* __restrict__ Kh,   // [4096][1024]
            const unsigned short* __restrict__ Vt,   // [32][64][2048]
            const unsigned char*  __restrict__ mask, // [2][2048][2048]
            unsigned short* __restrict__ ctx,        // [4096][1024]
            float* __restrict__ attn_out)            // [32][2048][2048]
{
    __shared__ __align__(16) unsigned short kbuf[2][16 * KROW]; // TDM-padded K
    __shared__ __align__(16) unsigned short pbuf[4][16][40];    // per-wave P

    const int tid  = threadIdx.x;
    const int lane = tid & 31;
    const int widx = tid >> 5;
    const int lr   = lane & 15;
    const int hi   = lane >> 4;

    const int bid = blockIdx.x;          // 1024 blocks
    const int b   = bid >> 9;
    const int rem = bid & 511;
    const int h   = rem >> 5;
    const int qt  = rem & 31;
    const int q0  = qt * 64 + widx * 16;

    const unsigned short* ksrc0 = Kh + (size_t)(b * 2048) * 1024 + h * 64;

    // Q A-fragments for d=0..63 (reused across all 128 key tiles)
    Frag16 a0, a1;
    {
        const unsigned short* qp =
            Qh + (size_t)(b * 2048 + q0 + lr) * 1024 + h * 64;
        a0.q[0] = *(const uint4*)(qp + hi * 16);
        a0.q[1] = *(const uint4*)(qp + hi * 16 + 8);
        a1.q[0] = *(const uint4*)(qp + 32 + hi * 16);
        a1.q[1] = *(const uint4*)(qp + 32 + hi * 16 + 8);
    }

    const float scale = 0.125f;          // 1/sqrt(64)
    const unsigned char* mbase = mask + (size_t)b * 2048 * 2048;

    float m[8], l[8];
    #pragma unroll
    for (int i = 0; i < 8; ++i) { m[i] = -3.0e38f; l[i] = 0.0f; }

    // ----------------------- pass 1: softmax stats ------------------------
#if ATHENA_HAVE_TDM
    if (widx == 0)
        tdm_issue_k_tile(ksrc0, (unsigned)(uintptr_t)&kbuf[0][0]);
#endif
    for (int kt = 0; kt < 128; ++kt) {
#if ATHENA_HAVE_TDM
        if (widx == 0) {
            if (kt + 1 < 128) {
                tdm_issue_k_tile(ksrc0 + (size_t)(kt + 1) * 16 * 1024,
                                 (unsigned)(uintptr_t)&kbuf[(kt + 1) & 1][0]);
                tdm_wait<1>();
            } else {
                tdm_wait<0>();
            }
        }
#else
        stage_k_tile_coop(&kbuf[kt & 1][0], ksrc0 + (size_t)kt * 16 * 1024, tid);
#endif
        __syncthreads();

        const unsigned short* krow = &kbuf[kt & 1][lr * KROW];
        Frag16 b0, b1;
        b0.q[0] = *(const uint4*)(krow + hi * 16);
        b0.q[1] = *(const uint4*)(krow + hi * 16 + 8);
        b1.q[0] = *(const uint4*)(krow + 32 + hi * 16);
        b1.q[1] = *(const uint4*)(krow + 32 + hi * 16 + 8);

        v8f c;
        #pragma unroll
        for (int e = 0; e < 8; ++e) c[e] = 0.0f;
        c = WMMA_BF16(a0.v, b0.v, c);
        c = WMMA_BF16(a1.v, b1.v, c);

        const int key = kt * 16 + lr;
        #pragma unroll
        for (int i = 0; i < 8; ++i) {
            const int row = q0 + i + hi * 8;
            float s = c[i] * scale;
            if (mbase[(size_t)row * 2048 + key]) s = -__builtin_inff();
            const float tm = half16_max(s);
            const float mn = fmaxf(m[i], tm);
            const float e  = half16_sum(__expf(s - mn));
            l[i] = l[i] * __expf(m[i] - mn) + e;
            m[i] = mn;
        }
        __syncthreads();
    }

    float invl[8];
    #pragma unroll
    for (int i = 0; i < 8; ++i) invl[i] = 1.0f / l[i];

    v8f o[4];
    #pragma unroll
    for (int j = 0; j < 4; ++j)
        #pragma unroll
        for (int e = 0; e < 8; ++e) o[j][e] = 0.0f;

    const size_t abase = (size_t)(h * 2 + b) * 2048 * 2048;
    const unsigned short* vhead = Vt + (size_t)(b * 16 + h) * 64 * 2048;

    // ------------------- pass 2: probs out + P @ V ------------------------
#if ATHENA_HAVE_TDM
    if (widx == 0)
        tdm_issue_k_tile(ksrc0, (unsigned)(uintptr_t)&kbuf[0][0]);
#endif
    for (int kt = 0; kt < 128; ++kt) {
#if ATHENA_HAVE_TDM
        if (widx == 0) {
            if (kt + 1 < 128) {
                tdm_issue_k_tile(ksrc0 + (size_t)(kt + 1) * 16 * 1024,
                                 (unsigned)(uintptr_t)&kbuf[(kt + 1) & 1][0]);
                tdm_wait<1>();
            } else {
                tdm_wait<0>();
            }
        }
#else
        stage_k_tile_coop(&kbuf[kt & 1][0], ksrc0 + (size_t)kt * 16 * 1024, tid);
#endif
        __syncthreads();

        const unsigned short* krow = &kbuf[kt & 1][lr * KROW];
        Frag16 b0, b1;
        b0.q[0] = *(const uint4*)(krow + hi * 16);
        b0.q[1] = *(const uint4*)(krow + hi * 16 + 8);
        b1.q[0] = *(const uint4*)(krow + 32 + hi * 16);
        b1.q[1] = *(const uint4*)(krow + 32 + hi * 16 + 8);

        v8f c;
        #pragma unroll
        for (int e = 0; e < 8; ++e) c[e] = 0.0f;
        c = WMMA_BF16(a0.v, b0.v, c);
        c = WMMA_BF16(a1.v, b1.v, c);

        const int key = kt * 16 + lr;
        #pragma unroll
        for (int i = 0; i < 8; ++i) {
            const int row = q0 + i + hi * 8;
            float s = c[i] * scale;
            if (mbase[(size_t)row * 2048 + key]) s = -__builtin_inff();
            const float p = __expf(s - m[i]) * invl[i];
            attn_out[abase + (size_t)row * 2048 + key] = p;       // fp32 probs
            pbuf[widx][i + hi * 8][(kt & 1) * 16 + lr] = f32_to_bf16(p);
        }

        if (kt & 1) {   // 32 keys staged -> one K-step of P @ V
            asm volatile("s_wait_dscnt 0" ::: "memory");  // lockstep wave: P ready
            Frag16 pa;  // A layout from LDS (row lr, K contiguous 16*hi..16*hi+15)
            pa.q[0] = *(const uint4*)(&pbuf[widx][lr][hi * 16]);
            pa.q[1] = *(const uint4*)(&pbuf[widx][lr][hi * 16 + 8]);
            const int kb = (kt - 1) * 16;
            #pragma unroll
            for (int j = 0; j < 4; ++j) {
                Frag16 vb;  // Vt row = d (contiguous keys) -> B layout direct
                const unsigned short* vrow =
                    vhead + (size_t)(j * 16 + lr) * 2048 + kb + hi * 16;
                vb.q[0] = *(const uint4*)(vrow);
                vb.q[1] = *(const uint4*)(vrow + 8);
                o[j] = WMMA_BF16(pa.v, vb.v, o[j]);
            }
        }
        __syncthreads();
    }

    // context out: [token][h*64 + d] bf16
    #pragma unroll
    for (int j = 0; j < 4; ++j)
        #pragma unroll
        for (int i = 0; i < 8; ++i) {
            const int row = b * 2048 + q0 + i + hi * 8;
            ctx[(size_t)row * 1024 + h * 64 + j * 16 + lr] = f32_to_bf16(o[j][i]);
        }
}

// ---------------------------------------------------------------------------
// Kernel 3: out = LayerNorm(ctx @ Wo^T + bo + residual) * gamma + beta
// Block = 16 rows x 1024 cols; wave w owns cols [w*128, w*128+128).
// Row mean/var: half-wave shuffle reduce + tiny LDS exchange across waves.
// ---------------------------------------------------------------------------
__global__ void __launch_bounds__(256)
outln_kernel(const unsigned short* __restrict__ ctx,
             const unsigned short* __restrict__ Wo,
             const float* __restrict__ bo,
             const float* __restrict__ residual,
             const float* __restrict__ gamma,
             const float* __restrict__ beta,
             float* __restrict__ out)
{
    __shared__ float s_sum[8][16];
    __shared__ float s_sq[8][16];

    const int lane = threadIdx.x & 31;
    const int widx = threadIdx.x >> 5;
    const int lr   = lane & 15;
    const int hi   = lane >> 4;
    const int m0   = blockIdx.x * 16;
    const int n0   = widx * 128;

    v8f acc[8];
    #pragma unroll
    for (int j = 0; j < 8; ++j)
        #pragma unroll
        for (int e = 0; e < 8; ++e) acc[j][e] = 0.0f;

    const unsigned short* arow = ctx + (size_t)(m0 + lr) * 1024;

    Frag16 a[2];
    auto loadA = [&](int buf, int kk) {
        a[buf].q[0] = *(const uint4*)(arow + kk + hi * 16);
        a[buf].q[1] = *(const uint4*)(arow + kk + hi * 16 + 8);
    };
    loadA(0, 0);
    #pragma unroll 2
    for (int kk = 0; kk < 1024; kk += 32) {
        const int cur = (kk >> 5) & 1;
        if (kk + 32 < 1024) loadA(cur ^ 1, kk + 32);
        #pragma unroll
        for (int j = 0; j < 8; ++j) {
            Frag16 bfr;
            const unsigned short* brow =
                Wo + (size_t)(n0 + j * 16 + lr) * 1024 + kk + hi * 16;
            bfr.q[0] = *(const uint4*)(brow);
            bfr.q[1] = *(const uint4*)(brow + 8);
            acc[j] = WMMA_BF16(a[cur].v, bfr.v, acc[j]);
        }
    }

    float ps[8], psq[8];
    #pragma unroll
    for (int i = 0; i < 8; ++i) { ps[i] = 0.0f; psq[i] = 0.0f; }

    #pragma unroll
    for (int j = 0; j < 8; ++j) {
        const int col = n0 + j * 16 + lr;
        const float bv = bo[col];
        #pragma unroll
        for (int i = 0; i < 8; ++i) {
            const int row = m0 + i + hi * 8;
            const float v = acc[j][i] + bv + residual[(size_t)row * 1024 + col];
            acc[j][i] = v;
            ps[i] += v;
            psq[i] += v * v;
        }
    }

    #pragma unroll
    for (int i = 0; i < 8; ++i) {
        const float su = half16_sum(ps[i]);
        const float sq = half16_sum(psq[i]);
        if (lr == 0) {
            s_sum[widx][i + hi * 8] = su;
            s_sq[widx][i + hi * 8]  = sq;
        }
    }
    __syncthreads();

    float mu[8], rstd[8];
    #pragma unroll
    for (int i = 0; i < 8; ++i) {
        const int rowl = i + hi * 8;
        float su = 0.0f, sq = 0.0f;
        #pragma unroll
        for (int w = 0; w < 8; ++w) { su += s_sum[w][rowl]; sq += s_sq[w][rowl]; }
        const float mval = su * (1.0f / 1024.0f);
        const float var  = sq * (1.0f / 1024.0f) - mval * mval;
        mu[i]   = mval;
        rstd[i] = rsqrtf(var + 1e-5f);
    }

    #pragma unroll
    for (int j = 0; j < 8; ++j) {
        const int col = n0 + j * 16 + lr;
        const float g = gamma[col];
        const float be = beta[col];
        #pragma unroll
        for (int i = 0; i < 8; ++i) {
            const int row = m0 + i + hi * 8;
            out[(size_t)row * 1024 + col] = (acc[j][i] - mu[i]) * rstd[i] * g + be;
        }
    }
}

// ---------------------------------------------------------------------------
// Host launcher
// ---------------------------------------------------------------------------
extern "C" void kernel_launch(void* const* d_in, const int* in_sizes, int n_in,
                              void* d_out, int out_size, void* d_ws, size_t ws_size,
                              hipStream_t stream) {
    const float* q     = (const float*)d_in[0];
    const float* k     = (const float*)d_in[1];
    const float* v     = (const float*)d_in[2];
    const unsigned char* mask = (const unsigned char*)d_in[3];
    const float* Wq    = (const float*)d_in[4];
    const float* bq    = (const float*)d_in[5];
    const float* Wk    = (const float*)d_in[6];
    const float* bk    = (const float*)d_in[7];
    const float* Wv    = (const float*)d_in[8];
    const float* bv    = (const float*)d_in[9];
    const float* Wo    = (const float*)d_in[10];
    const float* bo    = (const float*)d_in[11];
    const float* gamma = (const float*)d_in[12];
    const float* beta  = (const float*)d_in[13];

    float* out_ln   = (float*)d_out;                       // [2,2048,1024]
    float* out_attn = (float*)d_out + (size_t)4096 * 1024; // [32,2048,2048]

    const size_t ACT = (size_t)4096 * 1024;   // activation elements
    const size_t WEL = (size_t)1024 * 1024;   // weight elements

    unsigned short* wsb = (unsigned short*)d_ws;
    unsigned short* qb  = wsb;                 // bf16 activations
    unsigned short* kb  = qb  + ACT;
    unsigned short* vb  = kb  + ACT;
    unsigned short* wqb = vb  + ACT;           // bf16 weights
    unsigned short* wkb = wqb + WEL;
    unsigned short* wvb = wkb + WEL;
    unsigned short* wob = wvb + WEL;
    unsigned short* Qh  = wob + WEL;           // projected heads
    unsigned short* Kh  = Qh  + ACT;
    unsigned short* Vt  = Kh  + ACT;           // V transposed [32][64][2048]
    unsigned short* ctx = Vt  + ACT;           // attention context

    // 1) fp32 -> bf16 conversions
    cvt_bf16_kernel<<<(int)((ACT + 255) / 256), 256, 0, stream>>>(q, qb, (int)ACT);
    cvt_bf16_kernel<<<(int)((ACT + 255) / 256), 256, 0, stream>>>(k, kb, (int)ACT);
    cvt_bf16_kernel<<<(int)((ACT + 255) / 256), 256, 0, stream>>>(v, vb, (int)ACT);
    cvt_bf16_kernel<<<(int)((WEL + 255) / 256), 256, 0, stream>>>(Wq, wqb, (int)WEL);
    cvt_bf16_kernel<<<(int)((WEL + 255) / 256), 256, 0, stream>>>(Wk, wkb, (int)WEL);
    cvt_bf16_kernel<<<(int)((WEL + 255) / 256), 256, 0, stream>>>(Wv, wvb, (int)WEL);
    cvt_bf16_kernel<<<(int)((WEL + 255) / 256), 256, 0, stream>>>(Wo, wob, (int)WEL);

    // 2) Q/K/V projections (WMMA bf16, register double-buffered)
    proj_gemm_kernel<<<1024, 128, 0, stream>>>(qb, wqb, bq, Qh, 0);
    proj_gemm_kernel<<<1024, 128, 0, stream>>>(kb, wkb, bk, Kh, 0);
    proj_gemm_kernel<<<1024, 128, 0, stream>>>(vb, wvb, bv, Vt, 1);

    // 3) attention (TDM-staged K tiles, two-pass softmax, WMMA QK^T and P@V)
    attn_kernel<<<1024, 128, 0, stream>>>(Qh, Kh, Vt, mask, ctx, out_attn);

    // 4) output projection + residual + LayerNorm
    outln_kernel<<<256, 256, 0, stream>>>(ctx, wob, bo, q, gamma, beta, out_ln);
}
